// KRNN_61495341744281
// MI455X (gfx1250) — matrix operand order repaired and checked by reading, not compiled, not run on hardware
//
#include <hip/hip_runtime.h>
#include <hip/hip_bf16.h>

typedef __attribute__((ext_vector_type(16))) __bf16 v16bf;
typedef __attribute__((ext_vector_type(8)))  float  v8f;

// ---------------- problem constants ----------------
constexpr int kB = 32, kN = 207, kTin = 24, kTout = 12, kF = 2, kOV = 3, kH = 64, kP = 10;
constexpr int kInEnc = kF * kOV + 4;   // 10
constexpr int kInDec = kOV + 4;        // 7
constexpr int kBN    = kB * kN;        // 6624
constexpr int kTiles = kBN / 16;       // 414 (exact)
constexpr int kG     = 3 * kH;         // 192 gate columns (r|z|n)
constexpr int kBiasK = 10;             // constant-1.0 column carrying bih in the GEMM

// ---------------- LDS layout (bytes); total 157.2 KB -> 2 blocks per 320 KB WGP ----
constexpr int ENC_OFF = 0;                          // float [24][16][64]  mixed encoder outputs
constexpr int XIN_OFF = ENC_OFF + kTin*16*kH*4;     // bf16  [24][16][16]  inputs (K 16..31 implicit 0)
constexpr int WFR_OFF = XIN_OFF + kTin*16*16*2;     // bf16  36 B-fragments x 512 elems
constexpr int B2_OFF  = WFR_OFF + 36*1024;          // float [192] bhh (bih folded into K=10 row)
constexpr int HST_OFF = B2_OFF + kG*4;              // bf16  [16][64] h (A-fragment source)
constexpr int HF_OFF  = HST_OFF + 16*kH*2;          // float [16][64] mixed hidden / decoder h
constexpr int AW_OFF  = HF_OFF + 16*kH*4;           // float [16][10] per-row ensemble weights
constexpr int SC_OFF  = AW_OFF + 16*kP*4;           // float [24][16] attention scores/weights
constexpr int LST_OFF = SC_OFF + kTin*16*4;         // float [16][3]  autoregressive lst
constexpr int XD_OFF  = LST_OFF + 16*3*4;           // bf16  [16][16] decoder xin
constexpr int SMEM_BYTES = XD_OFF + 16*16*2;        // 157248 B

// ---------------- fast transcendentals on the recurrent critical path ----------------
__device__ inline float fast_rcp(float x) {
#if __has_builtin(__builtin_amdgcn_rcpf)
  return __builtin_amdgcn_rcpf(x);
#else
  return 1.0f / x;
#endif
}
__device__ inline float fast_tanh(float x) {
#if __has_builtin(__builtin_amdgcn_tanhf)
  return __builtin_amdgcn_tanhf(x);          // gfx1250 V_TANH_F32 (TRANS pipe)
#elif __has_builtin(__builtin_amdgcn_tanh_f32)
  return __builtin_amdgcn_tanh_f32(x);
#else
  const float e = __expf(-2.0f * x);
  return (1.0f - e) * fast_rcp(1.0f + e);
#endif
}
__device__ inline float fsig(float x) { return fast_rcp(1.0f + __expf(-x)); }

// ---------------- fragment loaders ----------------
__device__ inline v16bf ld_frag2(const __bf16* p0, const __bf16* p1) {
  union { v16bf v; uint4 q[2]; } u;
  u.q[0] = *(const uint4*)p0;   // elements 0..7  -> VGPRs 0..3 (K pairs)
  u.q[1] = *(const uint4*)p1;   // elements 8..15 -> VGPRs 4..7
  return u.v;
}
// B fragment: pre-swizzled contiguous 32B per lane in LDS.
__device__ inline v16bf ld_bfrag(const __bf16* wf, int frag, int lane) {
  const __bf16* p = wf + frag*512 + lane*16;
  return ld_frag2(p, p + 8);
}
// A fragment from full 64-wide row-major region (h), 32-col window at `win`.
__device__ inline v16bf ld_afrag(const __bf16* base, int stride, int lane, int win) {
  const int m  = lane & 15;
  const int kb = (lane >> 4) * 8;
  const __bf16* r = base + m*stride + win + kb;
  return ld_frag2(r, r + 16);
}
// A fragment whose K=16..31 half is structurally zero (inputs stored 16 wide).
__device__ inline v16bf ld_afrag_half(const __bf16* base, int lane) {
  const int m = lane & 15, kb = (lane >> 4) * 8;
  union { v16bf v; uint4 q[2]; } u;
  u.q[0] = *(const uint4*)(base + m*16 + kb);
  u.q[1] = make_uint4(0u, 0u, 0u, 0u);
  return u.v;
}
__device__ inline v8f wmma_bf16(v16bf a, v16bf b, v8f c) {
  return __builtin_amdgcn_wmma_f32_16x16x32_bf16(false, a, false, b, (short)0, c, false, false);
}

// One GRU cell for a 16-row tile. Gates r|z|n across 192 cols (12 N-tiles).
// G1 = x@Wih^T + bih (bias via K=10 constant-1 column, C inits to inline 0),
// G2 = h@Whh^T + bhh (K-chunks 1,2). Updates hc in place.
__device__ inline void gru_step(const __bf16* wf, const float* b2,
                                v16bf a0, v16bf a1, v16bf a2, v8f hc[4], int lane) {
  const int ncol = lane & 15;
  #pragma unroll
  for (int i = 0; i < 4; ++i) {
    v8f g1[3], g2[3];
    #pragma unroll
    for (int g = 0; g < 3; ++g) {          // g: 0=r, 1=z, 2=n
      const int nt = g*4 + i;
      const float bb2 = b2[nt*16 + ncol];
      v8f c1 = {};                         // zero -> SRC2 inline 0
      v8f c2;
      #pragma unroll
      for (int j = 0; j < 8; ++j) c2[j] = bb2;
      g1[g] = wmma_bf16(a0, ld_bfrag(wf,  0 + nt, lane), c1);
      c2    = wmma_bf16(a1, ld_bfrag(wf, 12 + nt, lane), c2);
      g2[g] = wmma_bf16(a2, ld_bfrag(wf, 24 + nt, lane), c2);
    }
    #pragma unroll
    for (int j = 0; j < 8; ++j) {
      float r = fsig(g1[0][j] + g2[0][j]);
      float z = fsig(g1[1][j] + g2[1][j]);
      float n = fast_tanh(g1[2][j] + r * g2[2][j]);
      hc[i][j] = (1.0f - z) * n + z * hc[i][j];
    }
  }
}

// ---------------- fused kernel: 1 wave == 16 batch rows, full seq2seq ----------------
__global__ void __launch_bounds__(32)
krnn_fused(const float* __restrict__ X,       const float* __restrict__ enc_pos,
           const float* __restrict__ enc_Wih, const float* __restrict__ enc_Whh,
           const float* __restrict__ enc_bih, const float* __restrict__ enc_bhh,
           const float* __restrict__ attn,    const float* __restrict__ dec_pos,
           const float* __restrict__ dec_Wih, const float* __restrict__ dec_Whh,
           const float* __restrict__ dec_bih, const float* __restrict__ dec_bhh,
           const float* __restrict__ lin_W,   const float* __restrict__ lin_b,
           float* __restrict__ out)
{
  extern __shared__ char smem[];
  float*  encAcc = (float*) (smem + ENC_OFF);
  __bf16* xinSeq = (__bf16*)(smem + XIN_OFF);
  __bf16* wfrag  = (__bf16*)(smem + WFR_OFF);
  float*  b2     = (float*) (smem + B2_OFF);
  __bf16* hstage = (__bf16*)(smem + HST_OFF);
  float*  hf     = (float*) (smem + HF_OFF);
  float*  aw     = (float*) (smem + AW_OFF);
  float*  sc     = (float*) (smem + SC_OFF);
  float*  lst    = (float*) (smem + LST_OFF);
  __bf16* xinDec = (__bf16*)(smem + XD_OFF);

  const int lane = threadIdx.x;
  const int row0 = blockIdx.x * 16;

  // ---- init scratch
  for (int i = lane; i < kTin*16*kH; i += 32) encAcc[i] = 0.0f;
  for (int i = lane; i < 16*kH;     i += 32) hf[i]     = 0.0f;
  for (int i = lane; i < 16*16;     i += 32)
    xinDec[i] = ((i & 15) == kBiasK) ? (__bf16)1.0f : (__bf16)0.0f;

  // ---- per-row ensemble softmax aw[n,:] and autoregressive seed lst
  if (lane < 16) {
    const int row = row0 + lane;
    const int n   = row % kN;
    float v[kP], mx = -1e30f;
    for (int p = 0; p < kP; ++p) { v[p] = attn[n*kP + p]; mx = fmaxf(mx, v[p]); }
    float s = 0.0f;
    for (int p = 0; p < kP; ++p) { v[p] = __expf(v[p] - mx); s += v[p]; }
    const float inv = fast_rcp(s);
    for (int p = 0; p < kP; ++p) aw[lane*kP + p] = v[p] * inv;
    for (int rep = 0; rep < kOV; ++rep)
      lst[lane*3 + rep] = X[(row*kTin + (kTin-1-rep))*kF + 0];
  }

  // ---- overlapped inputs (16 K-cols): 0..5 = [x_t,x_{t-1},x_{t-2}], 6..9 pos (per p),
  //      10 = 1.0 (bias column), 11..15 = 0; K 16..31 implicit zero at fragment load.
  for (int idx = lane; idx < kTin*16*16; idx += 32) {
    const int k = idx & 15, m = (idx >> 4) & 15, t = idx >> 8;
    float val = 0.0f;
    if (k < 6) {
      const int rep = k >> 1, f = k & 1, ts = t - rep;
      if (ts >= 0) val = X[((row0 + m)*kTin + ts)*kF + f];
    } else if (k == kBiasK) val = 1.0f;
    xinSeq[idx] = (__bf16)val;
  }
  __syncthreads();

  // =================== encoders: fuse ensemble mix into encAcc/hf ===================
  for (int p = 0; p < kP; ++p) {
    // position columns for this encoder
    for (int idx = lane; idx < kTin*4; idx += 32) {
      const int t = idx >> 2, c = idx & 3;
      const float v = enc_pos[(p*kTin + t)*4 + c];
      for (int m = 0; m < 16; ++m) xinSeq[(t*16 + m)*16 + 6 + c] = (__bf16)v;
    }
    // weights -> B-fragment layout: [Wih | bih@k=10 | pad | Whh], K padded to 96
    for (int frag = 0; frag < 36; ++frag) {
      const int kc = frag / 12, nt = frag % 12;
      const int nl = lane & 15, kg = (lane >> 4) * 8;
      const int n  = nt*16 + nl;
      __bf16* dst = wfrag + frag*512 + lane*16;
      #pragma unroll
      for (int v = 0; v < 8; ++v)
        #pragma unroll
        for (int h2 = 0; h2 < 2; ++h2) {
          const int k = kc*32 + kg + 2*v + h2 + (v >= 4 ? 8 : 0);
          float w = 0.0f;
          if (k < kInEnc)        w = enc_Wih[(p*kG + n)*kInEnc + k];
          else if (k == kBiasK)  w = enc_bih[p*kG + n];
          else if (k >= 32)      w = enc_Whh[(p*kG + n)*kH + (k - 32)];
          dst[2*v + h2] = (__bf16)w;
        }
    }
    for (int i = lane; i < kG;    i += 32) b2[i] = enc_bhh[p*kG + i];
    for (int i = lane; i < 16*kH; i += 32) hstage[i] = (__bf16)0.0f;

    float awv[8];
    const int mb = 8 * (lane >> 4);
    #pragma unroll
    for (int j = 0; j < 8; ++j) awv[j] = aw[(j + mb)*kP + p];

    v8f hc[4];
    #pragma unroll
    for (int i = 0; i < 4; ++i)
      #pragma unroll
      for (int j = 0; j < 8; ++j) hc[i][j] = 0.0f;
    __syncthreads();

    const int ncol = lane & 15;
    for (int t = 0; t < kTin; ++t) {
      v16bf a0 = ld_afrag_half(xinSeq + t*16*16, lane);
      v16bf a1 = ld_afrag(hstage, kH, lane, 0);
      v16bf a2 = ld_afrag(hstage, kH, lane, 32);
      gru_step(wfrag, b2, a0, a1, a2, hc, lane);

      #pragma unroll
      for (int i = 0; i < 4; ++i)
        #pragma unroll
        for (int j = 0; j < 8; ++j) {
          const float hn = hc[i][j];
          const int m = j + mb, col = i*16 + ncol;
          hstage[m*kH + col] = (__bf16)hn;
          encAcc[(t*16 + m)*kH + col] += awv[j] * hn;   // softmax-mixed on the fly
        }
      __syncthreads();
    }
    // mixed final hidden state (register-resident h_T)
    #pragma unroll
    for (int i = 0; i < 4; ++i)
      #pragma unroll
      for (int j = 0; j < 8; ++j)
        hf[(j + mb)*kH + i*16 + ncol] += awv[j] * hc[i][j];
    __syncthreads();
  }

  // =================== decoder ===================
  for (int frag = 0; frag < 36; ++frag) {
    const int kc = frag / 12, nt = frag % 12;
    const int nl = lane & 15, kg = (lane >> 4) * 8;
    const int n  = nt*16 + nl;
    __bf16* dst = wfrag + frag*512 + lane*16;
    #pragma unroll
    for (int v = 0; v < 8; ++v)
      #pragma unroll
      for (int h2 = 0; h2 < 2; ++h2) {
        const int k = kc*32 + kg + 2*v + h2 + (v >= 4 ? 8 : 0);
        float w = 0.0f;
        if (k < kInDec)        w = dec_Wih[n*kInDec + k];
        else if (k == kBiasK)  w = dec_bih[n];
        else if (k >= 32)      w = dec_Whh[n*kH + (k - 32)];
        dst[2*v + h2] = (__bf16)w;
      }
  }
  for (int i = lane; i < kG; i += 32) b2[i] = dec_bhh[i];
  __syncthreads();

  const float lwb = lin_b[0];
  const int ncol = lane & 15, mb = 8 * (lane >> 4);
  for (int t = 0; t < kTout; ++t) {
    // attention scores against current h
    for (int e = lane; e < kTin*16; e += 32) {
      const int tt = e >> 4, m = e & 15;
      float d = 0.0f;
      for (int k = 0; k < kH; ++k) d += encAcc[(tt*16 + m)*kH + k] * hf[m*kH + k];
      sc[tt*16 + m] = d;
    }
    __syncthreads();
    if (lane < 16) {                       // softmax over time per row
      float mx = -1e30f;
      for (int tt = 0; tt < kTin; ++tt) mx = fmaxf(mx, sc[tt*16 + lane]);
      float e_[kTin], s = 0.0f;
      for (int tt = 0; tt < kTin; ++tt) { e_[tt] = __expf(sc[tt*16 + lane] - mx); s += e_[tt]; }
      const float inv = fast_rcp(s);
      for (int tt = 0; tt < kTin; ++tt) sc[tt*16 + lane] = e_[tt] * inv;
    }
    __syncthreads();
    // h += ctx ; stage bf16 copy for WMMA A-side
    for (int e = lane; e < 16*kH; e += 32) {
      const int m = e >> 6, k = e & 63;
      float c = 0.0f;
      for (int tt = 0; tt < kTin; ++tt) c += sc[tt*16 + m] * encAcc[(tt*16 + m)*kH + k];
      const float nh = hf[e] + c;
      hf[e] = nh;
      hstage[e] = (__bf16)nh;
    }
    // decoder input: [lst(3) | pos_t(4)]; col 10 stays 1.0 (bias), rest zero
    for (int e = lane; e < 16*kInDec; e += 32) {
      const int m = e / kInDec, c = e % kInDec;
      const float v = (c < 3) ? lst[m*3 + c] : dec_pos[t*4 + (c - 3)];
      xinDec[m*16 + c] = (__bf16)v;
    }
    __syncthreads();

    v16bf a0 = ld_afrag_half(xinDec, lane);
    v16bf a1 = ld_afrag(hstage, kH, lane, 0);
    v16bf a2 = ld_afrag(hstage, kH, lane, 32);
    v8f hc[4];
    #pragma unroll
    for (int i = 0; i < 4; ++i)
      #pragma unroll
      for (int j = 0; j < 8; ++j) hc[i][j] = hf[(j + mb)*kH + i*16 + ncol];

    gru_step(wfrag, b2, a0, a1, a2, hc, lane);

    #pragma unroll
    for (int i = 0; i < 4; ++i)
      #pragma unroll
      for (int j = 0; j < 8; ++j) hf[(j + mb)*kH + i*16 + ncol] = hc[i][j];
    __syncthreads();

    if (lane < 16) {                       // linear head + autoregressive feedback
      float d = lwb;
      for (int k = 0; k < kH; ++k) d += hf[lane*kH + k] * lin_W[k];
      out[(row0 + lane)*kTout + t] = d;
      lst[lane*3 + 0] = d;
    }
    __syncthreads();
  }
}

// ---------------- host launch ----------------
extern "C" void kernel_launch(void* const* d_in, const int* in_sizes, int n_in,
                              void* d_out, int out_size, void* d_ws, size_t ws_size,
                              hipStream_t stream) {
  (void)in_sizes; (void)n_in; (void)d_ws; (void)ws_size; (void)out_size;
  // d_in order: A(0, unused), X(1), enc_pos(2), enc_Wih(3), enc_Whh(4), enc_bih(5),
  // enc_bhh(6), attn(7), dec_pos(8), dec_Wih(9), dec_Whh(10), dec_bih(11),
  // dec_bhh(12), lin_W(13), lin_b(14)
  const float* X       = (const float*)d_in[1];
  const float* enc_pos = (const float*)d_in[2];
  const float* enc_Wih = (const float*)d_in[3];
  const float* enc_Whh = (const float*)d_in[4];
  const float* enc_bih = (const float*)d_in[5];
  const float* enc_bhh = (const float*)d_in[6];
  const float* attn    = (const float*)d_in[7];
  const float* dec_pos = (const float*)d_in[8];
  const float* dec_Wih = (const float*)d_in[9];
  const float* dec_Whh = (const float*)d_in[10];
  const float* dec_bih = (const float*)d_in[11];
  const float* dec_bhh = (const float*)d_in[12];
  const float* lin_W   = (const float*)d_in[13];
  const float* lin_b   = (const float*)d_in[14];

  krnn_fused<<<dim3(kTiles), dim3(32), SMEM_BYTES, stream>>>(
      X, enc_pos, enc_Wih, enc_Whh, enc_bih, enc_bhh, attn,
      dec_pos, dec_Wih, dec_Whh, dec_bih, dec_bhh, lin_W, lin_b,
      (float*)d_out);
}
